// GNS_heterogeneous_33243046871576
// MI455X (gfx1250) — compile-verified, matrix operand order checked by reading, not run on hardware
//
#include <hip/hip_runtime.h>
#include <hip/hip_bf16.h>
#include <stdint.h>

#define NBUS 50000
#define NGEN 10000
#define EBB  300000
#define HID  32
#define HEADS 4
#define HD   128
#define NLAYERS 3
#define IN_BUS 11
#define IN_GEN 3
#define EDIM 4
#define OUT_BUS 5
#define OUT_GEN 1

typedef __attribute__((ext_vector_type(16))) _Float16 v16h;
typedef __attribute__((ext_vector_type(8)))  _Float16 v8h;
typedef __attribute__((ext_vector_type(8)))  float    v8f;

// ---------------- device helpers ----------------
__device__ __forceinline__ float lrelu(float x){ return x > 0.f ? x : 0.01f*x; }

__device__ __forceinline__ float wsum(float x){
  #pragma unroll
  for (int o = 16; o > 0; o >>= 1) x += __shfl_xor(x, o, 32);
  return x;
}

// monotone float <-> uint encoding for atomicMax on floats
__device__ __forceinline__ unsigned enc_f(float f){
  unsigned u = __float_as_uint(f);
  return (u & 0x80000000u) ? ~u : (u | 0x80000000u);
}
__device__ __forceinline__ float dec_f(unsigned e){
  unsigned u = (e & 0x80000000u) ? (e & 0x7fffffffu) : ~e;
  return __uint_as_float(u);
}
#define ENC_NEG_INF 0x007FFFFFu   // enc_f(-inf)

// ---------------- utility kernels ----------------
__global__ void k_fill_f32(float* __restrict__ p, float v, int n){
  int i = blockIdx.x*blockDim.x + threadIdx.x;
  if (i < n) p[i] = v;
}
__global__ void k_fill_u32(unsigned* __restrict__ p, unsigned v, int n){
  int i = blockIdx.x*blockDim.x + threadIdx.x;
  if (i < n) p[i] = v;
}

// Convert W[din x dout] (f32, row-major) into WMMA B-fragment layout:
// Wf[ kb ][ n ][ kw ] , kb = k/32, kw = k%32  -> each column's 32-deep K block
// is contiguous, so a lane grabs its fragment with two b128 loads.
__global__ void k_w_frag(const float* __restrict__ W, _Float16* __restrict__ Wf,
                         int din, int dout){
  int idx = blockIdx.x*blockDim.x + threadIdx.x;
  if (idx >= din*dout) return;
  int k = idx / dout, n = idx % dout;
  Wf[((size_t)(k >> 5)*dout + n)*32 + (k & 31)] = (_Float16)W[idx];
}

// ---------------- WMMA GEMM ----------------
// out[NxDout] = X[NxDin](f32 -> f16) @ W(frag f16) + bias.
// N%16==0, din%32==0, dout%(16*CT)==0. One wave computes a 16 x (16*CT) strip,
// reusing its A fragment across CT v_wmma issues per K step.
template<int CT>
__global__ void k_lin_wmma(const float* __restrict__ X,
                           const _Float16* __restrict__ Wf,
                           const float* __restrict__ bias,
                           float* __restrict__ out,
                           int N, int din, int dout){
  const int lane  = threadIdx.x & 31;
  const int wib   = threadIdx.x >> 5;
  const int wpb   = blockDim.x >> 5;
  const int gwave = blockIdx.x * wpb + wib;
  const int nwave = gridDim.x * wpb;
  const int groupsM = dout / (16*CT);
  const int totalTiles = (N >> 4) * groupsM;
  const int m  = lane & 15;
  const int hf = lane >> 4;

  for (int tile = gwave; tile < totalTiles; tile += nwave){
    const int tn = tile / groupsM;
    const int tg = tile % groupsM;
    const int r0 = tn << 4, c0 = tg * (16*CT);

    v8f acc[CT];
    #pragma unroll
    for (int ct = 0; ct < CT; ++ct){
      float bc = bias[c0 + ct*16 + m];
      #pragma unroll
      for (int i = 0; i < 8; ++i) acc[ct][i] = bc;
    }

    for (int kk = 0; kk < din; kk += 32){
      const float* xr = X + (size_t)(r0 + m) * din + kk;
      v8f xlo = *(const v8f*)(xr + hf*8);
      v8f xhi = *(const v8f*)(xr + 16 + hf*8);
      v8h alo = __builtin_convertvector(xlo, v8h);
      v8h ahi = __builtin_convertvector(xhi, v8h);
      v16h a = __builtin_shufflevector(alo, ahi,
                 0,1,2,3,4,5,6,7,8,9,10,11,12,13,14,15);

      const _Float16* wblk = Wf + ((size_t)(kk >> 5) * dout) * 32;
      __builtin_prefetch(wblk + (size_t)dout*32, 0, 1);
      #pragma unroll
      for (int ct = 0; ct < CT; ++ct){
        const _Float16* colp = wblk + (size_t)(c0 + ct*16 + m) * 32;
        v8h blo = *(const v8h*)(colp + hf*8);
        v8h bhi = *(const v8h*)(colp + 16 + hf*8);
        v16h b = __builtin_shufflevector(blo, bhi,
                   0,1,2,3,4,5,6,7,8,9,10,11,12,13,14,15);
        acc[ct] = __builtin_amdgcn_wmma_f32_16x16x32_f16(false, a, false, b,
                                                         (short)0, acc[ct],
                                                         false, false);
      }
    }
    #pragma unroll
    for (int ct = 0; ct < CT; ++ct){
      float* orow = out + (size_t)r0 * dout + c0 + ct*16 + m;
      #pragma unroll
      for (int v = 0; v < 8; ++v)
        orow[(size_t)(v + hf*8) * dout] = acc[ct][v];
    }
  }
}

// ---------------- input MLP proj: LN( leaky(x@W1+b1) @ W2 + b2 ) ----------------
// one wave per row, lane = hidden channel (HID==32==warpSize)
__global__ void k_proj(const float* __restrict__ X, int din,
                       const float* __restrict__ W1, const float* __restrict__ b1,
                       const float* __restrict__ W2, const float* __restrict__ b2,
                       const float* __restrict__ g,  const float* __restrict__ bn,
                       float* __restrict__ out, int N){
  int row = (blockIdx.x*blockDim.x + threadIdx.x) >> 5;
  int j = threadIdx.x & 31;
  if (row >= N) return;
  const float* x = X + (size_t)row * din;
  float h = b1[j];
  for (int i = 0; i < din; ++i) h += x[i] * W1[i*HID + j];
  h = lrelu(h);
  float h2 = b2[j];
  #pragma unroll
  for (int k = 0; k < HID; ++k) h2 += __shfl(h, k, 32) * W2[k*HID + j];
  float mean = wsum(h2) * (1.f/HID);
  float d = h2 - mean;
  float var = wsum(d*d) * (1.f/HID);
  out[(size_t)row*HID + j] = d * rsqrtf(var + 1e-5f) * g[j] + bn[j];
}

// ---------------- attention: logits + running segment max ----------------
__global__ void k_att_logit(int E, const int* __restrict__ src, const int* __restrict__ dst,
                            const float* __restrict__ q, const float* __restrict__ k,
                            const float* __restrict__ ep,
                            float* __restrict__ a, unsigned* __restrict__ mord){
  int idx = blockIdx.x*blockDim.x + threadIdx.x;
  if (idx >= E*HEADS) return;
  int e = idx >> 2, h = idx & 3;
  int s = src[e], d = dst[e];
  const float* qp = q + (size_t)d*HD + h*HID;
  const float* kp = k + (size_t)s*HD + h*HID;
  float acc = 0.f;
  if (ep){
    const float* pp = ep + (size_t)e*HD + h*HID;
    for (int i = 0; i < HID; ++i) acc += qp[i] * (kp[i] + pp[i]);
  } else {
    for (int i = 0; i < HID; ++i) acc += qp[i] * kp[i];
  }
  acc *= 0.17677669529663687f;   // 1/sqrt(32)
  a[idx] = acc;
  atomicMax(&mord[(size_t)d*HEADS + h], enc_f(acc));
}

__global__ void k_exp_sum(int E, const int* __restrict__ dst,
                          float* __restrict__ a, const unsigned* __restrict__ mord,
                          float* __restrict__ ssum){
  int idx = blockIdx.x*blockDim.x + threadIdx.x;
  if (idx >= E*HEADS) return;
  int e = idx >> 2, h = idx & 3;
  int d = dst[e];
  float ex = __expf(a[idx] - dec_f(mord[(size_t)d*HEADS + h]));
  a[idx] = ex;
  atomicAdd(&ssum[(size_t)d*HEADS + h], ex);
}

__global__ void k_scatter(int E, const int* __restrict__ src, const int* __restrict__ dst,
                          const float* __restrict__ v, const float* __restrict__ ep,
                          const float* __restrict__ a, const float* __restrict__ ssum,
                          float* __restrict__ out){
  int t = blockIdx.x*blockDim.x + threadIdx.x;
  if (t >= E*HD) return;
  int e = t >> 7, c = t & 127, h = c >> 5;
  int d = dst[e], s = src[e];
  float w = a[(size_t)e*HEADS + h] / ssum[(size_t)d*HEADS + h];
  float val = v[(size_t)s*HD + c];
  if (ep) val += ep[(size_t)e*HD + c];
  atomicAdd(&out[(size_t)d*HD + c], val * w);
}

// ---------------- gated skip: beta*xr + (1-beta)*out ----------------
__global__ void k_gate(const float* __restrict__ conv, const float* __restrict__ skip,
                       const float* __restrict__ B, float* __restrict__ dst,
                       int N, int accum){
  int row = (blockIdx.x*blockDim.x + threadIdx.x) >> 5;
  int lane = threadIdx.x & 31;
  if (row >= N) return;
  const float* o  = conv + (size_t)row*HD;
  const float* xr = skip + (size_t)row*HD;
  float part = 0.f;
  #pragma unroll
  for (int i = 0; i < 4; ++i){
    int c = lane + 32*i;
    float ov = o[c], xv = xr[c];
    part += ov*B[c] + xv*B[HD+c] + (ov - xv)*B[2*HD + c];
  }
  float t = wsum(part);
  float beta = 1.f / (1.f + __expf(-t));
  float* dr = dst + (size_t)row*HD;
  #pragma unroll
  for (int i = 0; i < 4; ++i){
    int c = lane + 32*i;
    float val = beta*xr[c] + (1.f - beta)*o[c];
    dr[c] = accum ? dr[c] + val : val;
  }
}

// ---------------- leaky(LN(x)) (+residual) ----------------
__global__ void k_ln_leaky_res(const float* __restrict__ x,
                               const float* __restrict__ g, const float* __restrict__ b,
                               float* __restrict__ h, int N, int add){
  int row = (blockIdx.x*blockDim.x + threadIdx.x) >> 5;
  int lane = threadIdx.x & 31;
  if (row >= N) return;
  const float* xr = x + (size_t)row*HD;
  float v0[4]; float s = 0.f;
  #pragma unroll
  for (int i = 0; i < 4; ++i){ v0[i] = xr[lane + 32*i]; s += v0[i]; }
  float mean = wsum(s) * (1.f/HD);
  float sq = 0.f;
  #pragma unroll
  for (int i = 0; i < 4; ++i){ float d = v0[i]-mean; sq += d*d; }
  float rs = rsqrtf(wsum(sq)*(1.f/HD) + 1e-5f);
  float* hr = h + (size_t)row*HD;
  #pragma unroll
  for (int i = 0; i < 4; ++i){
    int c = lane + 32*i;
    float y = lrelu((v0[i]-mean)*rs*g[c] + b[c]);
    hr[c] = add ? hr[c] + y : y;
  }
}

// ---------------- head tail: LN -> leaky -> small linear -> masked select ----------------
__global__ void k_head_finish(const float* __restrict__ hid,
                              const float* __restrict__ g, const float* __restrict__ bn,
                              const float* __restrict__ W2, const float* __restrict__ b2,
                              const unsigned char* __restrict__ mask,
                              const float* __restrict__ fixed, int fstride,
                              int dof, float* __restrict__ out, int N){
  int row = (blockIdx.x*blockDim.x + threadIdx.x) >> 5;
  int j = threadIdx.x & 31;
  if (row >= N) return;
  float h = hid[(size_t)row*HID + j];
  float mean = wsum(h) * (1.f/HID);
  float d = h - mean;
  float var = wsum(d*d) * (1.f/HID);
  float t = lrelu(d * rsqrtf(var + 1e-5f) * g[j] + bn[j]);
  int jj = j < dof ? j : 0;
  float acc = b2[jj];
  #pragma unroll
  for (int k = 0; k < HID; ++k) acc += __shfl(t, k, 32) * W2[k*dof + jj];
  if (j < dof){
    float fv = fixed[(size_t)row*fstride + j];
    out[(size_t)row*dof + j] = mask[(size_t)row*dof + j] ? acc : fv;
  }
}

// ---------------- physics ----------------
__global__ void k_branch_flow(const int* __restrict__ ei, const float* __restrict__ bt,
                              const float* __restrict__ ea,
                              float* __restrict__ Pin, float* __restrict__ Qin, int E){
  int e = blockIdx.x*blockDim.x + threadIdx.x;
  if (e >= E) return;
  int f = ei[e], t = ei[E + e];
  float vmf = bt[(size_t)f*OUT_BUS + 3], vaf = bt[(size_t)f*OUT_BUS + 4];
  float vmt = bt[(size_t)t*OUT_BUS + 3], vat = bt[(size_t)t*OUT_BUS + 4];
  float vfr = vmf*cosf(vaf), vfi = vmf*sinf(vaf);
  float vtr = vmt*cosf(vat), vti = vmt*sinf(vat);
  float yffr = ea[(size_t)e*EDIM+0], yffi = ea[(size_t)e*EDIM+1];
  float yftr = ea[(size_t)e*EDIM+2], yfti = ea[(size_t)e*EDIM+3];
  float Ir = yffr*vfr - yffi*vfi + yftr*vtr - yfti*vti;
  float Ii = yffr*vfi + yffi*vfr + yftr*vti + yfti*vtr;
  atomicAdd(&Pin[f], vfr*Ir + vfi*Ii);
  atomicAdd(&Qin[f], vfi*Ir - vfr*Ii);
}
__global__ void k_agg(const int* __restrict__ gbd, const float* __restrict__ gt,
                      float* __restrict__ agg, int n){
  int i = blockIdx.x*blockDim.x + threadIdx.x;
  if (i < n) atomicAdd(&agg[gbd[i]], gt[i]);
}
__global__ void k_resid(const float* __restrict__ bt, const float* __restrict__ xb,
                        const float* __restrict__ agg,
                        const float* __restrict__ Pin, const float* __restrict__ Qin,
                        float* __restrict__ rP, float* __restrict__ rQ, int n){
  int b = blockIdx.x*blockDim.x + threadIdx.x;
  if (b >= n) return;
  float vm = bt[(size_t)b*OUT_BUS + 3]; float vm2 = vm*vm;
  rP[b] = agg[b] - bt[(size_t)b*OUT_BUS + 0] - xb[(size_t)b*IN_BUS + 5]*vm2 - Pin[b];
  rQ[b] = bt[(size_t)b*OUT_BUS + 2] - bt[(size_t)b*OUT_BUS + 1] + xb[(size_t)b*IN_BUS + 6]*vm2 - Qin[b];
}
__global__ void k_phys(float* __restrict__ h, const float* __restrict__ rP,
                       const float* __restrict__ rQ,
                       const float* __restrict__ W, const float* __restrict__ b, int n){
  int t = blockIdx.x*blockDim.x + threadIdx.x;
  if (t >= n) return;
  int bi = t >> 7, c = t & 127;
  h[t] += lrelu(rP[bi]*W[c] + rQ[bi]*W[HD + c] + b[c]);
}
__global__ void k_output(const float* __restrict__ bt, const float* __restrict__ rP,
                         const float* __restrict__ rQ, const float* __restrict__ gt,
                         float* __restrict__ out){
  int i = blockIdx.x*blockDim.x + threadIdx.x;
  if (i < NBUS){
    #pragma unroll
    for (int j = 0; j < OUT_BUS; ++j) out[(size_t)i*7 + j] = bt[(size_t)i*OUT_BUS + j];
    out[(size_t)i*7 + 5] = rP[i];
    out[(size_t)i*7 + 6] = rQ[i];
  } else if (i < NBUS + NGEN){
    out[(size_t)NBUS*7 + (i - NBUS)] = gt[i - NBUS];
  }
}

// ---------------- host-side parameter offsets (jax tree_flatten order: sorted keys) ----
struct OffLin { int W, b; };
struct OffConv { int beta; OffLin e; OffLin k, q, skip, v; int has_e; };
struct OffLayer { OffConv bb, bg, gb; int lnb_b, lnb_g, lng_b, lng_g; };
struct OffAll {
  OffLayer L[NLAYERS];
  int mb_bn, mb_g; OffLin mb_l1, mb_l2;
  int mg_bn, mg_g; OffLin mg_l1, mg_l2;
  int pb_bn, pb_g; OffLin pb_l1, pb_l2;
  int pe_bn, pe_g; OffLin pe_l1, pe_l2;
  int pg_bn, pg_g; OffLin pg_l1, pg_l2;
  OffLin phys;
  int total, nleaves;
  int leaf_off[200], leaf_sz[200];
};
static int leafo(OffAll& o, int& cur, int n){
  int off = cur; o.leaf_off[o.nleaves] = off; o.leaf_sz[o.nleaves] = n;
  o.nleaves++; cur += n; return off;
}
static void mk_lin(OffAll& o, int& cur, OffLin& l, int di, int dn){
  l.W = leafo(o, cur, di*dn); l.b = leafo(o, cur, dn);
}
static void mk_conv(OffAll& o, int& cur, OffConv& c, int din, int edge){
  c.beta = leafo(o, cur, 3*HD);
  c.has_e = edge;
  if (edge) mk_lin(o, cur, c.e, HID, HD);
  mk_lin(o, cur, c.k, din, HD);
  mk_lin(o, cur, c.q, din, HD);
  mk_lin(o, cur, c.skip, din, HD);
  mk_lin(o, cur, c.v, din, HD);
}
static void mk_layer(OffAll& o, int& cur, OffLayer& L, int din){
  mk_conv(o, cur, L.bb, din, 1);
  mk_conv(o, cur, L.bg, din, 0);
  mk_conv(o, cur, L.gb, din, 0);
  L.lnb_b = leafo(o, cur, HD); L.lnb_g = leafo(o, cur, HD);
  L.lng_b = leafo(o, cur, HD); L.lng_g = leafo(o, cur, HD);
}
static OffAll make_offsets(){
  OffAll o{}; int cur = 0; o.nleaves = 0;
  mk_layer(o, cur, o.L[0], HID);
  mk_layer(o, cur, o.L[1], HD);
  mk_layer(o, cur, o.L[2], HD);
  o.mb_bn = leafo(o, cur, HID); o.mb_g = leafo(o, cur, HID);
  mk_lin(o, cur, o.mb_l1, HD, HID); mk_lin(o, cur, o.mb_l2, HID, OUT_BUS);
  o.mg_bn = leafo(o, cur, HID); o.mg_g = leafo(o, cur, HID);
  mk_lin(o, cur, o.mg_l1, HD, HID); mk_lin(o, cur, o.mg_l2, HID, OUT_GEN);
  o.pb_bn = leafo(o, cur, HID); o.pb_g = leafo(o, cur, HID);
  mk_lin(o, cur, o.pb_l1, IN_BUS, HID); mk_lin(o, cur, o.pb_l2, HID, HID);
  o.pe_bn = leafo(o, cur, HID); o.pe_g = leafo(o, cur, HID);
  mk_lin(o, cur, o.pe_l1, EDIM, HID); mk_lin(o, cur, o.pe_l2, HID, HID);
  o.pg_bn = leafo(o, cur, HID); o.pg_g = leafo(o, cur, HID);
  mk_lin(o, cur, o.pg_l1, IN_GEN, HID); mk_lin(o, cur, o.pg_l2, HID, HID);
  mk_lin(o, cur, o.phys, 2, HD);
  o.total = cur;
  return o;
}

static inline int cdiv(int a, int b){ return (a + b - 1) / b; }

extern "C" void kernel_launch(void* const* d_in, const int* in_sizes, int n_in,
                              void* d_out, int out_size, void* d_ws, size_t ws_size,
                              hipStream_t stream){
  (void)in_sizes; (void)out_size; (void)ws_size;
  const float* x_bus = (const float*)d_in[0];
  const float* x_gen = (const float*)d_in[1];
  const int*   ei_bb = (const int*)d_in[2];
  const float* ea_bb = (const float*)d_in[3];
  const int*   ei_gb = (const int*)d_in[4];
  const int*   ei_bg = (const int*)d_in[5];
  const unsigned char* mask_bus = (const unsigned char*)d_in[6];
  const unsigned char* mask_gen = (const unsigned char*)d_in[7];

  OffAll PO = make_offsets();

  // ---- workspace bump allocator ----
  char* base = (char*)d_ws; size_t cur = 0;
  auto allocB = [&](size_t bytes)->void*{
    void* p = base + cur; cur += (bytes + 255) & ~(size_t)255; return p;
  };
  auto allocF = [&](size_t n)->float*   { return (float*)allocB(n*sizeof(float)); };
  auto allocH = [&](size_t n)->_Float16*{ return (_Float16*)allocB(n*sizeof(_Float16)); };
  auto allocU = [&](size_t n)->unsigned*{ return (unsigned*)allocB(n*sizeof(unsigned)); };

  _Float16* Ph   = allocH(PO.total);      // fragment-layout f16 weights (same offsets as P)
  float*    Pbuf = allocF(PO.total);      // gathered params (if leaves come separately)
  float* h_bus32 = allocF((size_t)NBUS*HID);
  float* h_gen32 = allocF((size_t)NGEN*HID);
  float* e32     = allocF((size_t)EBB*HID);
  float* h_bus   = allocF((size_t)NBUS*HD);
  float* h_gen   = allocF((size_t)NGEN*HD);
  float* qb   = allocF((size_t)NBUS*HD);
  float* kb   = allocF((size_t)NBUS*HD);
  float* vb   = allocF((size_t)NBUS*HD);
  float* skb  = allocF((size_t)NBUS*HD);
  float* qg   = allocF((size_t)NGEN*HD);
  float* kg   = allocF((size_t)NGEN*HD);
  float* vg   = allocF((size_t)NGEN*HD);
  float* skg  = allocF((size_t)NGEN*HD);
  float* eproj= allocF((size_t)EBB*HD);
  float* att  = allocF((size_t)EBB*HEADS);
  unsigned* mord = allocU((size_t)NBUS*HEADS);
  float* ssum = allocF((size_t)NBUS*HEADS);
  float* ctmp = allocF((size_t)NBUS*HD);
  float* ob   = allocF((size_t)NBUS*HD);
  float* og   = allocF((size_t)NGEN*HD);
  float* hid32= allocF((size_t)NBUS*HID);
  float* bus_t= allocF((size_t)NBUS*OUT_BUS);
  float* gen_t= allocF((size_t)NGEN*OUT_GEN);
  float* Pin  = allocF(NBUS);
  float* Qin  = allocF(NBUS);
  float* agg  = allocF(NBUS);
  float* rP   = allocF(NBUS);
  float* rQ   = allocF(NBUS);

  // ---- resolve params buffer ----
  const float* P;
  if (n_in <= 9){
    P = (const float*)d_in[8];
  } else {
    for (int i = 0; i < PO.nleaves && 8 + i < n_in; ++i)
      hipMemcpyAsync(Pbuf + PO.leaf_off[i], d_in[8 + i],
                     (size_t)PO.leaf_sz[i]*sizeof(float),
                     hipMemcpyDeviceToDevice, stream);
    P = Pbuf;
  }

  // ---- convert all WMMA-consumed weights into fragment layout ----
  auto wfrag = [&](const OffLin& l, int din, int dout){
    k_w_frag<<<cdiv(din*dout,256),256,0,stream>>>(P + l.W, Ph + l.W, din, dout);
  };
  for (int i = 0; i < NLAYERS; ++i){
    const OffLayer& L = PO.L[i];
    int din = (i == 0) ? HID : HD;
    wfrag(L.bb.e, HID, HD);
    wfrag(L.bb.q, din, HD); wfrag(L.bb.k, din, HD);
    wfrag(L.bb.v, din, HD); wfrag(L.bb.skip, din, HD);
    wfrag(L.gb.q, din, HD); wfrag(L.gb.k, din, HD);
    wfrag(L.gb.v, din, HD); wfrag(L.gb.skip, din, HD);
    wfrag(L.bg.q, din, HD); wfrag(L.bg.k, din, HD);
    wfrag(L.bg.v, din, HD); wfrag(L.bg.skip, din, HD);
  }
  wfrag(PO.mb_l1, HD, HID);
  wfrag(PO.mg_l1, HD, HID);

  auto lin = [&](const float* X, const OffLin& l, float* out, int N, int din, int dout){
    if (dout == HD){
      int tiles = (N >> 4) * (dout / 64);
      int blocks = cdiv(tiles, 8); if (blocks > 4096) blocks = 4096;
      k_lin_wmma<4><<<blocks,256,0,stream>>>(X, Ph + l.W, P + l.b, out, N, din, dout);
    } else { // dout == 32
      int tiles = (N >> 4);
      int blocks = cdiv(tiles, 8); if (blocks > 4096) blocks = 4096;
      k_lin_wmma<2><<<blocks,256,0,stream>>>(X, Ph + l.W, P + l.b, out, N, din, dout);
    }
  };

  auto run_tconv = [&](const float* xsrc, const float* xdst, int nsrc, int ndst, int din,
                       const int* src, const int* dst, int E,
                       const OffConv& c, const float* ep,
                       float* outBuf, int accum,
                       float* qB, float* kB, float* vB, float* skB){
    lin(xdst, c.q,    qB,  ndst, din, HD);
    lin(xsrc, c.k,    kB,  nsrc, din, HD);
    lin(xsrc, c.v,    vB,  nsrc, din, HD);
    lin(xdst, c.skip, skB, ndst, din, HD);
    k_fill_u32<<<cdiv(ndst*HEADS,256),256,0,stream>>>(mord, ENC_NEG_INF, ndst*HEADS);
    k_fill_f32<<<cdiv(ndst*HEADS,256),256,0,stream>>>(ssum, 0.f, ndst*HEADS);
    k_fill_f32<<<cdiv(ndst*HD,256),256,0,stream>>>(ctmp, 0.f, ndst*HD);
    k_att_logit<<<cdiv(E*HEADS,256),256,0,stream>>>(E, src, dst, qB, kB, ep, att, mord);
    k_exp_sum<<<cdiv(E*HEADS,256),256,0,stream>>>(E, dst, att, mord, ssum);
    k_scatter<<<cdiv(E*HD,256),256,0,stream>>>(E, src, dst, vB, ep, att, ssum, ctmp);
    k_gate<<<cdiv(ndst,8),256,0,stream>>>(ctmp, skB, P + c.beta, outBuf, ndst, accum);
  };

  auto run_heads = [&](){
    lin(h_bus, PO.mb_l1, hid32, NBUS, HD, HID);
    k_head_finish<<<cdiv(NBUS,8),256,0,stream>>>(hid32, P+PO.mb_g, P+PO.mb_bn,
        P+PO.mb_l2.W, P+PO.mb_l2.b, mask_bus, x_bus, IN_BUS, OUT_BUS, bus_t, NBUS);
    lin(h_gen, PO.mg_l1, hid32, NGEN, HD, HID);
    k_head_finish<<<cdiv(NGEN,8),256,0,stream>>>(hid32, P+PO.mg_g, P+PO.mg_bn,
        P+PO.mg_l2.W, P+PO.mg_l2.b, mask_gen, x_gen, IN_GEN, OUT_GEN, gen_t, NGEN);
  };
  auto run_resid = [&](){
    k_fill_f32<<<cdiv(NBUS,256),256,0,stream>>>(Pin, 0.f, NBUS);
    k_fill_f32<<<cdiv(NBUS,256),256,0,stream>>>(Qin, 0.f, NBUS);
    k_fill_f32<<<cdiv(NBUS,256),256,0,stream>>>(agg, 0.f, NBUS);
    k_branch_flow<<<cdiv(EBB,256),256,0,stream>>>(ei_bb, bus_t, ea_bb, Pin, Qin, EBB);
    k_agg<<<cdiv(NGEN,256),256,0,stream>>>(ei_gb + NGEN, gen_t, agg, NGEN);
    k_resid<<<cdiv(NBUS,256),256,0,stream>>>(bus_t, x_bus, agg, Pin, Qin, rP, rQ, NBUS);
  };

  // ---- input projections ----
  k_proj<<<cdiv(NBUS,8),256,0,stream>>>(x_bus, IN_BUS,
      P+PO.pb_l1.W, P+PO.pb_l1.b, P+PO.pb_l2.W, P+PO.pb_l2.b,
      P+PO.pb_g, P+PO.pb_bn, h_bus32, NBUS);
  k_proj<<<cdiv(NGEN,8),256,0,stream>>>(x_gen, IN_GEN,
      P+PO.pg_l1.W, P+PO.pg_l1.b, P+PO.pg_l2.W, P+PO.pg_l2.b,
      P+PO.pg_g, P+PO.pg_bn, h_gen32, NGEN);
  k_proj<<<cdiv(EBB,8),256,0,stream>>>(ea_bb, EDIM,
      P+PO.pe_l1.W, P+PO.pe_l1.b, P+PO.pe_l2.W, P+PO.pe_l2.b,
      P+PO.pe_g, P+PO.pe_bn, e32, EBB);

  // ---- layers ----
  for (int i = 0; i < NLAYERS; ++i){
    const OffLayer& L = PO.L[i];
    int din = (i == 0) ? HID : HD;
    const float* hb_in = (i == 0) ? h_bus32 : h_bus;
    const float* hg_in = (i == 0) ? h_gen32 : h_gen;

    // bus<-bus (with edge features)
    lin(e32, L.bb.e, eproj, EBB, HID, HD);
    run_tconv(hb_in, hb_in, NBUS, NBUS, din, ei_bb, ei_bb + EBB, EBB,
              L.bb, eproj, ob, 0, qb, kb, vb, skb);
    // bus<-gen
    run_tconv(hg_in, hb_in, NGEN, NBUS, din, ei_gb, ei_gb + NGEN, NGEN,
              L.gb, nullptr, ob, 1, qb, kg, vg, skb);
    // gen<-bus
    run_tconv(hb_in, hg_in, NBUS, NGEN, din, ei_bg, ei_bg + NGEN, NGEN,
              L.bg, nullptr, og, 0, qg, kb, vb, skg);

    k_ln_leaky_res<<<cdiv(NBUS,8),256,0,stream>>>(ob, P+L.lnb_g, P+L.lnb_b, h_bus, NBUS, i > 0);
    k_ln_leaky_res<<<cdiv(NGEN,8),256,0,stream>>>(og, P+L.lng_g, P+L.lng_b, h_gen, NGEN, i > 0);

    run_heads();
    run_resid();
    k_phys<<<cdiv(NBUS*HD,256),256,0,stream>>>(h_bus, rP, rQ, P+PO.phys.W, P+PO.phys.b, NBUS*HD);
  }

  // ---- final heads, residuals, output ----
  run_heads();
  run_resid();
  k_output<<<cdiv(NBUS + NGEN,256),256,0,stream>>>(bus_t, rP, rQ, gen_t, (float*)d_out);
}
